// MultiHeadChannelAttention_11948599018094
// MI455X (gfx1250) — matrix operand order repaired and checked by reading, not compiled, not run on hardware
//
#include <hip/hip_runtime.h>
#include <hip/hip_bf16.h>

typedef __attribute__((ext_vector_type(16))) _Float16 v16h;
typedef __attribute__((ext_vector_type(8)))  float    v8f;

#define BATCH 2
#define CCH   512
#define NTOK  2048          // 64*32
#define HEADS 8
#define DK    64
#define MROWS (BATCH*NTOK)  // 4096

// ---------------- WMMA fragment loaders (wave32, 16x16x32 f16) ----------------
// A-fragment 16x32 (MxK): lane m = lane&15, half = lane>>4
//   element e -> k = 16*half + e
__device__ __forceinline__ v16h load_a_f16(const _Float16* src, int ld, int lane) {
  const int m = lane & 15, half = lane >> 4;
  const _Float16* row = src + (size_t)m * ld;
  v16h a;
#pragma unroll
  for (int e = 0; e < 16; ++e) {
    int k = 16 * half + e;
    a[e] = row[k];
  }
  return a;
}

// B-fragment 32x16 (KxN): lane col n = lane&15, element e -> k = 16*half + e.
// src points at (k=0, col=0); ldc = element stride between columns; the 16
// k-values per lane are CONTIGUOUS in memory (32B per lane -> 2x b128).
__device__ __forceinline__ v16h load_b_f16(const _Float16* src, int ldc, int lane) {
  const int n = lane & 15, half = lane >> 4;
  const _Float16* p = src + (size_t)n * ldc + 16 * half;
  v16h b;
#pragma unroll
  for (int e = 0; e < 16; ++e) b[e] = p[e];
  return b;
}

// ---------------- Kernel 0: one-shot f32 -> f16 conversion ----------------
__global__ void cvt_f16_kernel(const float* __restrict__ src,
                               _Float16* __restrict__ dst, int n) {
  int i = blockIdx.x * 256 + threadIdx.x;
  if (i < n) dst[i] = (_Float16)src[i];
}

// ---------------- Kernel 1: fused Q/K/V projection GEMMs ----------------
// Y = xt @ W^T + b ; per-wave 32 rows x 64 out-channels (B-frag reused x2).
// Q,K stored [b][h][n][d] (f16); V stored TRANSPOSED [b][h][d][n] (f16).
__global__ void qkv_proj_kernel(const _Float16* __restrict__ Xh,
                                const _Float16* __restrict__ Wqh,
                                const _Float16* __restrict__ Wkh,
                                const _Float16* __restrict__ Wvh,
                                const float* __restrict__ bq,
                                const float* __restrict__ bk,
                                const float* __restrict__ bv,
                                _Float16* __restrict__ Qw, _Float16* __restrict__ Kw,
                                _Float16* __restrict__ Vt) {
  const int lane = threadIdx.x;
  const int i0 = 32 * blockIdx.x;   // token row tile (global over B*N)
  const int c0 = 64 * blockIdx.y;   // output-channel tile
  const int mode = blockIdx.z;      // 0=Q 1=K 2=V

  const _Float16* W  = (mode == 0) ? Wqh : (mode == 1) ? Wkh : Wvh;
  const float* bias  = (mode == 0) ? bq  : (mode == 1) ? bk  : bv;

  v8f acc0[4] = {};
  v8f acc1[4] = {};
  const _Float16* A = Xh + (size_t)i0 * CCH;

  for (int k0 = 0; k0 < CCH; k0 += 32) {
    v16h af0 = load_a_f16(A + k0, CCH, lane);
    v16h af1 = load_a_f16(A + (size_t)16 * CCH + k0, CCH, lane);
#pragma unroll
    for (int t = 0; t < 4; ++t) {
      // B[k][n] = W[c0+16t+n][k] : k contiguous along W rows
      v16h bf = load_b_f16(W + (size_t)(c0 + 16 * t) * CCH + k0, CCH, lane);
      acc0[t] = __builtin_amdgcn_wmma_f32_16x16x32_f16(false, af0, false, bf,
                                                       (short)0, acc0[t], false, false);
      acc1[t] = __builtin_amdgcn_wmma_f32_16x16x32_f16(false, af1, false, bf,
                                                       (short)0, acc1[t], false, false);
    }
  }

  const int half = lane >> 4, ncol = lane & 15;
  const int b  = i0 / NTOK;
  const int n0 = i0 % NTOK;
#pragma unroll
  for (int t = 0; t < 4; ++t) {
    const int c = c0 + 16 * t + ncol;
    const int h = c >> 6, d = c & 63;
    const float bb = bias[c];
#pragma unroll
    for (int r = 0; r < 8; ++r) {
#pragma unroll
      for (int s = 0; s < 2; ++s) {   // row sub-tile
        const int nTok = n0 + 16 * s + r + 8 * half;
        const float av = (s == 0) ? acc0[t][r] : acc1[t][r];
        _Float16 val = (_Float16)(av + bb);
        if (mode == 2)
          Vt[(((size_t)b * HEADS + h) * DK + d) * NTOK + nTok] = val;
        else if (mode == 0)
          Qw[(((size_t)b * HEADS + h) * NTOK + nTok) * DK + d] = val;
        else
          Kw[(((size_t)b * HEADS + h) * NTOK + nTok) * DK + d] = val;
      }
    }
  }
}

// ---------------- Kernel 2: flash attention, one wave per 16-row tile ----------------
__global__ void flash_attn_kernel(const _Float16* __restrict__ Qw,
                                  const _Float16* __restrict__ Kw,
                                  const _Float16* __restrict__ Vt,
                                  _Float16* __restrict__ AO) {
  __shared__ _Float16 Pt[16 * 32];   // P-tile staging for D-layout -> A-layout
  const int lane = threadIdx.x;
  const int half = lane >> 4, ncol = lane & 15;
  const int i0 = 16 * blockIdx.x;    // query row tile within (b,h)
  const int bh = blockIdx.y;
  const int b  = bh / HEADS;
  const int h  = bh % HEADS;

  const _Float16* Qb = Qw + (size_t)bh * NTOK * DK;
  const _Float16* Kb = Kw + (size_t)bh * NTOK * DK;
  const _Float16* Vb = Vt + (size_t)bh * DK * NTOK;

  // Q tile 16x64 = two A-fragments (k 0..31, 32..63), resident in VGPRs
  v16h qf0 = load_a_f16(Qb + (size_t)i0 * DK, DK, lane);
  v16h qf1 = load_a_f16(Qb + (size_t)i0 * DK + 32, DK, lane);

  v8f O[4] = {};
  float mrow[8], lrow[8];
#pragma unroll
  for (int r = 0; r < 8; ++r) { mrow[r] = -1e30f; lrow[r] = 0.f; }

  const float scale = 0.125f;   // DK^-0.5

  for (int j0 = 0; j0 < NTOK; j0 += 32) {
    v8f s0 = {}, s1 = {};
    {
      v16h bk = load_b_f16(Kb + (size_t)j0 * DK, DK, lane);
      s0 = __builtin_amdgcn_wmma_f32_16x16x32_f16(false, qf0, false, bk, (short)0, s0, false, false);
      bk = load_b_f16(Kb + (size_t)j0 * DK + 32, DK, lane);
      s0 = __builtin_amdgcn_wmma_f32_16x16x32_f16(false, qf1, false, bk, (short)0, s0, false, false);
      bk = load_b_f16(Kb + (size_t)(j0 + 16) * DK, DK, lane);
      s1 = __builtin_amdgcn_wmma_f32_16x16x32_f16(false, qf0, false, bk, (short)0, s1, false, false);
      bk = load_b_f16(Kb + (size_t)(j0 + 16) * DK + 32, DK, lane);
      s1 = __builtin_amdgcn_wmma_f32_16x16x32_f16(false, qf1, false, bk, (short)0, s1, false, false);
    }

    // online softmax: row (r + 8*half) lives across the 16 lanes of this half,
    // so xor-shuffles 1/2/4/8 implement the row reduction without crossing halves.
#pragma unroll
    for (int r = 0; r < 8; ++r) {
      float a0 = s0[r] * scale, a1 = s1[r] * scale;
      float v = fmaxf(a0, a1);
      v = fmaxf(v, __shfl_xor(v, 1, 32));
      v = fmaxf(v, __shfl_xor(v, 2, 32));
      v = fmaxf(v, __shfl_xor(v, 4, 32));
      v = fmaxf(v, __shfl_xor(v, 8, 32));
      const float mnew = fmaxf(mrow[r], v);
      const float corr = __expf(mrow[r] - mnew);
      const float p0 = __expf(a0 - mnew);
      const float p1 = __expf(a1 - mnew);
      float ps = p0 + p1;
      ps += __shfl_xor(ps, 1, 32);
      ps += __shfl_xor(ps, 2, 32);
      ps += __shfl_xor(ps, 4, 32);
      ps += __shfl_xor(ps, 8, 32);
      lrow[r] = lrow[r] * corr + ps;
      mrow[r] = mnew;
      s0[r] = p0; s1[r] = p1;
#pragma unroll
      for (int t = 0; t < 4; ++t) O[t][r] *= corr;
    }

    // D-layout (row spread over lanes) -> A-layout (row per lane) via LDS
    __syncthreads();
#pragma unroll
    for (int r = 0; r < 8; ++r) {
      const int row = r + 8 * half;
      Pt[row * 32 + ncol]      = (_Float16)s0[r];
      Pt[row * 32 + 16 + ncol] = (_Float16)s1[r];
    }
    __syncthreads();
    v16h pf = load_a_f16(Pt, 32, lane);

    // O += P(16x32) * V(32x64); V stored transposed so B-frag lanes load 32B contiguous
#pragma unroll
    for (int t = 0; t < 4; ++t) {
      v16h vf = load_b_f16(Vb + (size_t)(16 * t) * NTOK + j0, NTOK, lane);
      O[t] = __builtin_amdgcn_wmma_f32_16x16x32_f16(false, pf, false, vf, (short)0, O[t], false, false);
    }
  }

  // finalize: divide by row sums, store f16 attn-out as (B*N, C) with c = h*64+d
#pragma unroll
  for (int r = 0; r < 8; ++r) {
    const int i = i0 + r + 8 * half;
    const float rinv = 1.f / lrow[r];
#pragma unroll
    for (int t = 0; t < 4; ++t) {
      const int d = 16 * t + ncol;
      AO[((size_t)b * NTOK + i) * CCH + h * DK + d] = (_Float16)(O[t][r] * rinv);
    }
  }
}

// ---------------- Kernel 3: output projection, f32 result ----------------
__global__ void fc_out_kernel(const _Float16* __restrict__ AO,
                              const _Float16* __restrict__ Wfch,
                              const float* __restrict__ bfc,
                              float* __restrict__ out) {
  const int lane = threadIdx.x;
  const int i0 = 32 * blockIdx.x;
  const int c0 = 64 * blockIdx.y;

  v8f acc0[4] = {};
  v8f acc1[4] = {};
  const _Float16* A = AO + (size_t)i0 * CCH;
  for (int k0 = 0; k0 < CCH; k0 += 32) {
    v16h af0 = load_a_f16(A + k0, CCH, lane);
    v16h af1 = load_a_f16(A + (size_t)16 * CCH + k0, CCH, lane);
#pragma unroll
    for (int t = 0; t < 4; ++t) {
      v16h bf = load_b_f16(Wfch + (size_t)(c0 + 16 * t) * CCH + k0, CCH, lane);
      acc0[t] = __builtin_amdgcn_wmma_f32_16x16x32_f16(false, af0, false, bf,
                                                       (short)0, acc0[t], false, false);
      acc1[t] = __builtin_amdgcn_wmma_f32_16x16x32_f16(false, af1, false, bf,
                                                       (short)0, acc1[t], false, false);
    }
  }
  const int half = lane >> 4, ncol = lane & 15;
#pragma unroll
  for (int t = 0; t < 4; ++t) {
    const int c = c0 + 16 * t + ncol;
    const float bb = bfc[c];
#pragma unroll
    for (int r = 0; r < 8; ++r) {
      out[(size_t)(i0 + r + 8 * half) * CCH + c]      = acc0[t][r] + bb;
      out[(size_t)(i0 + 16 + r + 8 * half) * CCH + c] = acc1[t][r] + bb;
    }
  }
}

extern "C" void kernel_launch(void* const* d_in, const int* in_sizes, int n_in,
                              void* d_out, int out_size, void* d_ws, size_t ws_size,
                              hipStream_t stream) {
  (void)in_sizes; (void)n_in; (void)out_size; (void)ws_size;
  const float* x   = (const float*)d_in[0];
  const float* Wq  = (const float*)d_in[1];
  const float* bq  = (const float*)d_in[2];
  const float* Wk  = (const float*)d_in[3];
  const float* bk  = (const float*)d_in[4];
  const float* Wv  = (const float*)d_in[5];
  const float* bv  = (const float*)d_in[6];
  const float* Wfc = (const float*)d_in[7];
  const float* bfc = (const float*)d_in[8];
  float* out = (float*)d_out;

  // ---- workspace layout (all f16) ----
  const size_t xElems = (size_t)MROWS * CCH;                // 2M
  const size_t wElems = (size_t)CCH * CCH;                  // 256K each
  const size_t perMat = (size_t)BATCH * HEADS * NTOK * DK;  // 2M each
  _Float16* Xh   = (_Float16*)d_ws;        // 4 MB
  _Float16* Wqh  = Xh  + xElems;           // 512 KB
  _Float16* Wkh  = Wqh + wElems;
  _Float16* Wvh  = Wkh + wElems;
  _Float16* Wfch = Wvh + wElems;
  _Float16* Qw   = Wfch + wElems;          // 4 MB
  _Float16* Kw   = Qw + perMat;            // 4 MB
  _Float16* Vt   = Kw + perMat;            // 4 MB
  _Float16* AO   = Vt + perMat;            // 4 MB   (total ~22 MB)

  // one-shot f32 -> f16 conversions (removes cvt work from all GEMM hot loops)
  cvt_f16_kernel<<<(int)((xElems + 255) / 256), 256, 0, stream>>>(x,   Xh,   (int)xElems);
  cvt_f16_kernel<<<(int)((wElems + 255) / 256), 256, 0, stream>>>(Wq,  Wqh,  (int)wElems);
  cvt_f16_kernel<<<(int)((wElems + 255) / 256), 256, 0, stream>>>(Wk,  Wkh,  (int)wElems);
  cvt_f16_kernel<<<(int)((wElems + 255) / 256), 256, 0, stream>>>(Wv,  Wvh,  (int)wElems);
  cvt_f16_kernel<<<(int)((wElems + 255) / 256), 256, 0, stream>>>(Wfc, Wfch, (int)wElems);

  dim3 g1(MROWS / 32, CCH / 64, 3);
  qkv_proj_kernel<<<g1, 32, 0, stream>>>(Xh, Wqh, Wkh, Wvh, bq, bk, bv, Qw, Kw, Vt);

  dim3 g2(NTOK / 16, BATCH * HEADS, 1);
  flash_attn_kernel<<<g2, 32, 0, stream>>>(Qw, Kw, Vt, AO);

  dim3 g3(MROWS / 32, CCH / 64, 1);
  fc_out_kernel<<<g3, 32, 0, stream>>>(AO, Wfch, bfc, out);
}
